// SelfAttention_Layer_74990128988635
// MI455X (gfx1250) — compile-verified, matrix operand order checked by reading, not run on hardware
//
#include <hip/hip_runtime.h>

#define Bd 16
#define Sd 2048
#define Hd 1024

typedef __attribute__((ext_vector_type(16))) __bf16 v16bf;
typedef __attribute__((ext_vector_type(8)))  float  v8f;

union FragU {
  unsigned int u[8];
  v16bf v;
};

__device__ __forceinline__ v8f vzero() {
  v8f z = {0.f, 0.f, 0.f, 0.f, 0.f, 0.f, 0.f, 0.f};
  return z;
}

// round-to-nearest-even float -> bf16 bits
__device__ __forceinline__ unsigned short f2bf(float x) {
  unsigned int u = __float_as_uint(x);
  u += 0x7FFFu + ((u >> 16) & 1u);
  return (unsigned short)(u >> 16);
}

// A/B fragment loader from a row-major bf16 (ushort) tile in LDS.
// Element (row, k) at base[row*ld + k].  Per ISA 7.12.2 16-bit A 16x32 layout:
// lane half (lane>=16) selects K+8 / K+24 groups; VGPR v holds K pair.
__device__ __forceinline__ v16bf frag_lds(const unsigned short* base, int rowBase,
                                          int ld, int kOfs, int lane) {
  const int half = (lane >> 4) & 1;
  const int r = rowBase + (lane & 15);
  FragU f;
#pragma unroll
  for (int v = 0; v < 8; ++v) {
    const int k = kOfs + ((v < 4) ? 0 : 16) + half * 8 + (v & 3) * 2;
    f.u[v] = *(const unsigned int*)(base + r * ld + k);
  }
  return f.v;
}

// Same fragment pattern, streaming from global bf16 row-major [rows][ld].
__device__ __forceinline__ v16bf frag_global(const unsigned short* base, long rowBase,
                                             long ld, int kOfs, int lane) {
  const int half = (lane >> 4) & 1;
  const long r = rowBase + (lane & 15);
  FragU f;
#pragma unroll
  for (int v = 0; v < 8; ++v) {
    const int k = kOfs + ((v < 4) ? 0 : 16) + half * 8 + (v & 3) * 2;
    f.u[v] = *(const unsigned int*)(base + r * ld + k);
  }
  return f.v;
}

__device__ __forceinline__ v8f wmma_bf16(v16bf a, v16bf b, v8f c) {
  return __builtin_amdgcn_wmma_f32_16x16x32_bf16(false, a, false, b, (short)0, c,
                                                 false, false);
}

// ---------------------------------------------------------------------------
// Kernel 1: Q/K/V = X @ W^T + b, output bf16.  M=B*S=32768, N=K=H=1024.
// grid = (M/16, N/128, 3), block = 256 (8 waves); wave w owns 16x16 tile at n=w*16.
// ---------------------------------------------------------------------------
__global__ __launch_bounds__(256) void qkv_proj_kernel(
    const float* __restrict__ X,
    const float* __restrict__ Wq, const float* __restrict__ bq,
    const float* __restrict__ Wk, const float* __restrict__ bk,
    const float* __restrict__ Wv, const float* __restrict__ bv,
    unsigned short* __restrict__ Qb, unsigned short* __restrict__ Kb,
    unsigned short* __restrict__ Vb) {
  __shared__ unsigned short Xs[16 * 128];    // 4 KB
  __shared__ unsigned short Ws[128 * 128];   // 32 KB

  const float* W;
  const float* bias;
  unsigned short* Out;
  if (blockIdx.z == 0)      { W = Wq; bias = bq; Out = Qb; }
  else if (blockIdx.z == 1) { W = Wk; bias = bk; Out = Kb; }
  else                      { W = Wv; bias = bv; Out = Vb; }

  const int t = threadIdx.x;
  const int lane = t & 31, wave = t >> 5;
  const long mbase = (long)blockIdx.x * 16;
  const int nbase = blockIdx.y * 128;

  v8f acc = vzero();

  for (int kc = 0; kc < 8; ++kc) {
    const int k0 = kc * 128;
    // stage X tile 16x128 (f32 -> bf16)
    {
      const int elem = t * 8;
      const int row = elem >> 7, col = elem & 127;
      const float* src = X + (mbase + row) * Hd + k0 + col;
      float4 q0 = *(const float4*)(src);
      float4 q1 = *(const float4*)(src + 4);
      unsigned short* dst = Xs + row * 128 + col;
      dst[0] = f2bf(q0.x); dst[1] = f2bf(q0.y); dst[2] = f2bf(q0.z); dst[3] = f2bf(q0.w);
      dst[4] = f2bf(q1.x); dst[5] = f2bf(q1.y); dst[6] = f2bf(q1.z); dst[7] = f2bf(q1.w);
    }
    // stage W tile 128x128 (f32 -> bf16); W is row-major [n][k] == B-fragment layout
    {
      const int row = t >> 1;
      const int colb = (t & 1) * 64;
      const float* src = W + (long)(nbase + row) * Hd + k0 + colb;
      unsigned short* dst = Ws + row * 128 + colb;
#pragma unroll
      for (int i = 0; i < 16; ++i) {
        float4 q = *(const float4*)(src + i * 4);
        dst[i * 4 + 0] = f2bf(q.x);
        dst[i * 4 + 1] = f2bf(q.y);
        dst[i * 4 + 2] = f2bf(q.z);
        dst[i * 4 + 3] = f2bf(q.w);
      }
    }
    __syncthreads();
#pragma unroll
    for (int kk = 0; kk < 4; ++kk) {
      v16bf a = frag_lds(Xs, 0, 128, kk * 32, lane);
      v16bf b = frag_lds(Ws, wave * 16, 128, kk * 32, lane);
      acc = wmma_bf16(a, b, acc);
    }
    __syncthreads();
  }

  // bias add + bf16 store.  C layout: VGPR r -> M = r + (lane>=16 ? 8 : 0), N = lane&15.
  const int nglob = nbase + wave * 16 + (lane & 15);
  const float bvad = bias[nglob];
  const int mofs = (lane >> 4) ? 8 : 0;
#pragma unroll
  for (int r = 0; r < 8; ++r) {
    const long mrow = mbase + r + mofs;
    Out[mrow * Hd + nglob] = f2bf(acc[r] + bvad);
  }
}

// ---------------------------------------------------------------------------
// Kernel 2: scores = (Q K^T)/sqrt(H), key-mask, softmax, P stored as bf16.
// grid = B * S/16 = 2048 q-tiles; 8 waves x 256 keys = full 2048-key row.
// ---------------------------------------------------------------------------
__global__ __launch_bounds__(256) void attn_scores_kernel(
    const unsigned short* __restrict__ Qb, const unsigned short* __restrict__ Kb,
    const int* __restrict__ mask, unsigned short* __restrict__ P) {
  __shared__ unsigned short Qs[16 * 1024];   // 32 KB bf16 Q strip
  __shared__ float red[16][8];
  __shared__ float rowmax[16];
  __shared__ float rowinv[16];

  const int t = threadIdx.x, lane = t & 31, wave = t >> 5;
  const int b = blockIdx.x >> 7;            // 128 q-tiles per batch
  const int qbase = (blockIdx.x & 127) * 16;
  const int rbase = (lane >> 4) ? 8 : 0;

  // stage Q strip (contiguous 32 KB) into LDS
  {
    const uint4* src = (const uint4*)(Qb + (long)(b * Sd + qbase) * Hd);
    uint4* dst = (uint4*)Qs;
    for (int i = t; i < (16 * 1024) / 8; i += 256) dst[i] = src[i];
  }
  __syncthreads();

  const long keyRow0 = (long)b * Sd + wave * 256;
  v8f accs[16];
#pragma unroll
  for (int nt = 0; nt < 16; ++nt) {
    v8f acc = vzero();
    // hint L2: prefetch this tile's key rows
    __builtin_prefetch(Kb + (keyRow0 + nt * 16 + (lane & 15)) * Hd, 0, 1);
    for (int kc = 0; kc < 32; ++kc) {
      v16bf a = frag_lds(Qs, 0, 1024, kc * 32, lane);
      v16bf bb = frag_global(Kb, keyRow0 + nt * 16, Hd, kc * 32, lane);
      acc = wmma_bf16(a, bb, acc);
    }
    accs[nt] = acc;
  }

  // scale + key mask + per-lane row maxima
  const float scale = 0.03125f;  // 1/sqrt(1024)
  float mymax[8];
#pragma unroll
  for (int r = 0; r < 8; ++r) mymax[r] = -3.0e38f;
#pragma unroll
  for (int nt = 0; nt < 16; ++nt) {
    const int key = wave * 256 + nt * 16 + (lane & 15);
    const bool valid = mask[b * Sd + key] != 0;
#pragma unroll
    for (int r = 0; r < 8; ++r) {
      const float s = valid ? accs[nt][r] * scale : -1.0e9f;
      accs[nt][r] = s;
      mymax[r] = fmaxf(mymax[r], s);
    }
  }
  // reduce max across the 16 lanes sharing each M row, then across waves via LDS
#pragma unroll
  for (int r = 0; r < 8; ++r) {
#pragma unroll
    for (int off = 8; off >= 1; off >>= 1)
      mymax[r] = fmaxf(mymax[r], __shfl_xor(mymax[r], off, 16));
  }
  if ((lane & 15) == 0) {
#pragma unroll
    for (int r = 0; r < 8; ++r) red[rbase + r][wave] = mymax[r];
  }
  __syncthreads();
  if (t < 16) {
    float mx = red[t][0];
#pragma unroll
    for (int w = 1; w < 8; ++w) mx = fmaxf(mx, red[t][w]);
    rowmax[t] = mx;
  }
  __syncthreads();

  // exp + per-lane partial sums
  float mysum[8];
#pragma unroll
  for (int r = 0; r < 8; ++r) mysum[r] = 0.f;
#pragma unroll
  for (int nt = 0; nt < 16; ++nt) {
#pragma unroll
    for (int r = 0; r < 8; ++r) {
      const float p = __expf(accs[nt][r] - rowmax[rbase + r]);
      accs[nt][r] = p;
      mysum[r] += p;
    }
  }
#pragma unroll
  for (int r = 0; r < 8; ++r) {
#pragma unroll
    for (int off = 8; off >= 1; off >>= 1)
      mysum[r] += __shfl_xor(mysum[r], off, 16);
  }
  __syncthreads();
  if ((lane & 15) == 0) {
#pragma unroll
    for (int r = 0; r < 8; ++r) red[rbase + r][wave] = mysum[r];
  }
  __syncthreads();
  if (t < 16) {
    float sm = 0.f;
#pragma unroll
    for (int w = 0; w < 8; ++w) sm += red[t][w];
    rowinv[t] = 1.0f / sm;
  }
  __syncthreads();

  // normalize and store P (bf16)
#pragma unroll
  for (int nt = 0; nt < 16; ++nt) {
    const int key = wave * 256 + nt * 16 + (lane & 15);
#pragma unroll
    for (int r = 0; r < 8; ++r) {
      const int m = rbase + r;
      const float p = accs[nt][r] * rowinv[m];
      P[(long)(b * Sd + qbase + m) * Sd + key] = f2bf(p);
    }
  }
}

// ---------------------------------------------------------------------------
// Kernel 3: O = P @ V, query-mask applied, fp32 output.
// grid = 2048 q-tiles; 8 waves x 128-wide H slices; V chunk transposed in LDS.
// ---------------------------------------------------------------------------
__global__ __launch_bounds__(256) void attn_out_kernel(
    const unsigned short* __restrict__ P, const unsigned short* __restrict__ Vb,
    const int* __restrict__ mask, float* __restrict__ Out) {
  __shared__ unsigned short Vt[32 * 1024];   // 64 KB, layout [h][key], ld = 32

  const int t = threadIdx.x, lane = t & 31, wave = t >> 5;
  const int b = blockIdx.x >> 7;
  const int qbase = (blockIdx.x & 127) * 16;

  v8f acc[8];
#pragma unroll
  for (int nt = 0; nt < 8; ++nt) acc[nt] = vzero();

  for (int kc = 0; kc < 64; ++kc) {
    // cooperative transpose-load of V chunk: 32 keys x 1024 h (bf16)
    {
      const unsigned int* src =
          (const unsigned int*)(Vb + (long)(b * Sd + kc * 32) * Hd);
      for (int i = t; i < 32 * 512; i += 256) {
        const unsigned int d = src[i];
        const int key = i >> 9;            // 512 dwords per key row
        const int h = (i & 511) * 2;
        Vt[h * 32 + key] = (unsigned short)(d & 0xffffu);
        Vt[(h + 1) * 32 + key] = (unsigned short)(d >> 16);
      }
    }
    __syncthreads();
    // A fragment: 16 q-rows x 32 keys of P, streamed from global (L2)
    v16bf a = frag_global(P, (long)(b * Sd + qbase), (long)Sd, kc * 32, lane);
#pragma unroll
    for (int nt = 0; nt < 8; ++nt) {
      v16bf bb = frag_lds(Vt, wave * 128 + nt * 16, 32, 0, lane);
      acc[nt] = wmma_bf16(a, bb, acc[nt]);
    }
    __syncthreads();
  }

  // store with query mask
  const int nl = lane & 15;
  const int mofs = (lane >> 4) ? 8 : 0;
#pragma unroll
  for (int nt = 0; nt < 8; ++nt) {
    const int h = wave * 128 + nt * 16 + nl;
#pragma unroll
    for (int r = 0; r < 8; ++r) {
      const int q = qbase + r + mofs;
      const float val = (mask[b * Sd + q] != 0) ? acc[nt][r] : 0.0f;
      Out[(long)(b * Sd + q) * Hd + h] = val;
    }
  }
}

// ---------------------------------------------------------------------------
extern "C" void kernel_launch(void* const* d_in, const int* in_sizes, int n_in,
                              void* d_out, int out_size, void* d_ws, size_t ws_size,
                              hipStream_t stream) {
  (void)in_sizes; (void)n_in; (void)out_size; (void)ws_size;

  const float* X    = (const float*)d_in[0];
  const int*   mask = (const int*)d_in[1];
  const float* Wq   = (const float*)d_in[2];
  const float* bq   = (const float*)d_in[3];
  const float* Wk   = (const float*)d_in[4];
  const float* bk   = (const float*)d_in[5];
  const float* Wv   = (const float*)d_in[6];
  const float* bv   = (const float*)d_in[7];
  float* Out = (float*)d_out;

  const size_t BSH = (size_t)Bd * Sd * Hd;  // 33,554,432
  unsigned short* Qb = (unsigned short*)d_ws;
  unsigned short* Kb = Qb + BSH;
  unsigned short* Vb = Kb + BSH;
  unsigned short* P  = Vb + BSH;            // B*S*S bf16 = 128 MB

  dim3 blk(256);
  qkv_proj_kernel<<<dim3((Bd * Sd) / 16, Hd / 128, 3), blk, 0, stream>>>(
      X, Wq, bq, Wk, bk, Wv, bv, Qb, Kb, Vb);
  attn_scores_kernel<<<dim3((Bd * Sd) / 16), blk, 0, stream>>>(Qb, Kb, mask, P);
  attn_out_kernel<<<dim3((Bd * Sd) / 16), blk, 0, stream>>>(P, Vb, mask, Out);
}
